// UnsupervisedModel_19911468384638
// MI455X (gfx1250) — matrix operand order repaired and checked by reading, not compiled, not run on hardware
//
#include <hip/hip_runtime.h>
#include <hip/hip_bf16.h>

typedef __attribute__((ext_vector_type(16))) _Float16 v16h;
typedef __attribute__((ext_vector_type(8)))  _Float16 v8h;
typedef __attribute__((ext_vector_type(8)))  float    v8f;

#define DIM      128
#define N_GRAPHS 64

// ---------------- utility kernels ----------------

__global__ void k_fill(float* __restrict__ p, float v, int n) {
    int i = blockIdx.x * blockDim.x + threadIdx.x;
    if (i < n) p[i] = v;
}

__global__ void k_deg_accum(const int* __restrict__ dst, const float* __restrict__ w,
                            float* __restrict__ deg, int nE) {
    int e = blockIdx.x * blockDim.x + threadIdx.x;
    if (e < nE) atomicAdd(&deg[dst[e]], w[e]);
}

__global__ void k_dinv(const float* __restrict__ deg, float* __restrict__ dinv, int n) {
    int i = blockIdx.x * blockDim.x + threadIdx.x;
    if (i < n) {
        float d = deg[i];
        dinv[i] = d > 0.f ? rsqrtf(d) : 0.f;
    }
}

__global__ void k_cvt_f16(const float* __restrict__ a, _Float16* __restrict__ b, int n) {
    int i = blockIdx.x * blockDim.x + threadIdx.x;
    if (i < n) b[i] = (_Float16)a[i];
}

// WT[n][k] = W[k][n], converted to f16 (128x128)
__global__ void k_cvt_wT(const float* __restrict__ W, _Float16* __restrict__ WT) {
    int i = blockIdx.x * blockDim.x + threadIdx.x;   // 0..16383
    int k = i >> 7, n = i & 127;
    WT[(size_t)n * DIM + k] = (_Float16)W[(size_t)k * DIM + n];
}

// ---------------- WMMA GEMM: C[M x 128] = A[M x 128] * B, with BT[n][k] in f16 ----------------
// One wave -> one 16x16 output tile. 8 waves/block cover a 16x128 row stripe.
__global__ void k_gemm_wmma(const _Float16* __restrict__ A, const _Float16* __restrict__ BT,
                            float* __restrict__ C, int M) {
    int lane = threadIdx.x & 31;
    int wave = threadIdx.x >> 5;       // 0..7 -> column tile n0 = wave*16
    int hl   = lane >> 4;              // half-wave selector
    int r    = lane & 15;
    int m0   = blockIdx.x * 16;
    if (m0 >= M) return;

    const _Float16* arow = A  + (size_t)(m0 + r) * DIM;
    const _Float16* brow = BT + (size_t)(wave * 16 + r) * DIM;

    v8f acc = {};
#pragma unroll
    for (int kk = 0; kk < DIM; kk += 32) {
        // A fragment (16x32 f16): lane holds K = hl*16 .. hl*16+15 (contiguous)
        v8h alo = *(const v8h*)(arow + kk + hl * 16);
        v8h ahi = *(const v8h*)(arow + kk + hl * 16 + 8);
        v16h a;
#pragma unroll
        for (int i = 0; i < 8; ++i) { a[i] = alo[i]; a[i + 8] = ahi[i]; }
        // B fragment (32x16 f16): lane holds col n0+r, K = hl*16 .. hl*16+15 (contiguous)
        v16h b = *(const v16h*)(brow + kk + hl * 16);
        acc = __builtin_amdgcn_wmma_f32_16x16x32_f16(false, a, false, b,
                                                     (short)0, acc, false, false);
    }

    // C/D layout: VGPR i -> row m0 + i + hl*8, col n0 + r
    float* cp = C + (size_t)m0 * DIM + wave * 16 + r;
#pragma unroll
    for (int i = 0; i < 8; ++i)
        cp[(size_t)(i + hl * 8) * DIM] = acc[i];
}

// ---------------- edge scatter: out[dst] += h[src] * norm ----------------
// one wave per edge; each lane handles 4 consecutive channels
__global__ void k_scatter(const int* __restrict__ src, const int* __restrict__ dst,
                          const float* __restrict__ ew, const float* __restrict__ dinv,
                          const float* __restrict__ h, float* __restrict__ out, int nE) {
    int t = blockIdx.x * blockDim.x + threadIdx.x;
    int e = t >> 5, lane = t & 31;
    if (e >= nE) return;
    int s = src[e], d = dst[e];
    float norm = dinv[s] * ew[e] * dinv[d];
    const float4 v = ((const float4*)(h + (size_t)s * DIM))[lane];
    float* o = out + (size_t)d * DIM + lane * 4;
    atomicAdd(o + 0, v.x * norm);
    atomicAdd(o + 1, v.y * norm);
    atomicAdd(o + 2, v.z * norm);
    atomicAdd(o + 3, v.w * norm);
}

// z = relu(out + h*dinv^2 + b) stored as f16 for next WMMA GEMM
__global__ void k_finalize_relu_f16(const float* __restrict__ out, const float* __restrict__ h,
                                    const float* __restrict__ dinv, const float* __restrict__ bias,
                                    _Float16* __restrict__ z, int n) {
    int i = blockIdx.x * blockDim.x + threadIdx.x;
    if (i >= n) return;
    int node = i >> 7, c = i & 127;
    float di = dinv[node];
    float v = out[i] + h[i] * di * di + bias[c];
    z[i] = (_Float16)fmaxf(v, 0.f);
}

// layer-2 finalize fused with graph pooling accumulation
__global__ void k_finalize_pool(const float* __restrict__ out, const float* __restrict__ h,
                                const float* __restrict__ dinv, const float* __restrict__ bias,
                                const int* __restrict__ batch, float* __restrict__ psum, int n) {
    int i = blockIdx.x * blockDim.x + threadIdx.x;
    if (i >= n) return;
    int node = i >> 7, c = i & 127;
    float di = dinv[node];
    float v = out[i] + h[i] * di * di + bias[c];
    atomicAdd(&psum[(size_t)batch[node] * DIM + c], v);
}

__global__ void k_count(const int* __restrict__ batch, float* __restrict__ cnt, int n) {
    int i = blockIdx.x * blockDim.x + threadIdx.x;
    if (i < n) atomicAdd(&cnt[batch[i]], 1.0f);
}

// out[g][n] = bfc[n] + sum_k (psum[g][k]/max(cnt[g],1)) * Wfc[k][n]
__global__ void k_fc(const float* __restrict__ psum, const float* __restrict__ cnt,
                     const float* __restrict__ Wfc, const float* __restrict__ bfc,
                     float* __restrict__ out) {
    int i = blockIdx.x * blockDim.x + threadIdx.x;   // 0..8191
    if (i >= N_GRAPHS * DIM) return;
    int g = i >> 7, no = i & 127;
    float inv = 1.f / fmaxf(cnt[g], 1.f);
    const float* row = psum + (size_t)g * DIM;
    float acc = bfc[no];
#pragma unroll 4
    for (int k = 0; k < DIM; ++k)
        acc += row[k] * inv * Wfc[(size_t)k * DIM + no];
    out[i] = acc;
}

// ---------------- launcher ----------------

extern "C" void kernel_launch(void* const* d_in, const int* in_sizes, int n_in,
                              void* d_out, int out_size, void* d_ws, size_t ws_size,
                              hipStream_t stream) {
    const float* x     = (const float*)d_in[0];
    const int*   ei    = (const int*)  d_in[1];
    const float* ew    = (const float*)d_in[2];
    const int*   batch = (const int*)  d_in[3];
    const float* W1    = (const float*)d_in[4];
    const float* b1    = (const float*)d_in[5];
    const float* W2    = (const float*)d_in[6];
    const float* b2    = (const float*)d_in[7];
    const float* Wfc   = (const float*)d_in[8];
    const float* bfc   = (const float*)d_in[9];

    const int N  = in_sizes[0] / DIM;   // 100000
    const int E  = in_sizes[1] / 2;     // 1600000
    const int ND = N * DIM;             // 12.8M
    const int* srcIdx = ei;
    const int* dstIdx = ei + E;

    // bump allocator in workspace (256B aligned)
    char* cur = (char*)d_ws;
    auto alloc = [&](size_t bytes) -> void* {
        void* p = (void*)cur;
        cur += (bytes + 255) & ~(size_t)255;
        return p;
    };
    float*    deg  = (float*)   alloc((size_t)N * 4);
    float*    dinv = (float*)   alloc((size_t)N * 4);
    _Float16* xh   = (_Float16*)alloc((size_t)ND * 2);       // reused as zh after layer 1
    _Float16* w1t  = (_Float16*)alloc((size_t)DIM * DIM * 2);
    _Float16* w2t  = (_Float16*)alloc((size_t)DIM * DIM * 2);
    float*    h    = (float*)   alloc((size_t)ND * 4);       // h1 then h2
    float*    accv = (float*)   alloc((size_t)ND * 4);       // scatter accumulator, reused
    float*    psum = (float*)   alloc((size_t)N_GRAPHS * DIM * 4);
    float*    cnt  = (float*)   alloc((size_t)N_GRAPHS * 4);

    const int T = 256;
    const int gN   = (N + T - 1) / T;
    const int gE   = (E + T - 1) / T;
    const int gND  = (ND + T - 1) / T;
    const int gEW  = (E * 32 + T - 1) / T;      // one wave per edge
    const int gW   = (DIM * DIM + T - 1) / T;
    const int gTile = N / 16;                   // 6250 row tiles

    // degrees (self-loop weight 1) and symmetric norm factors
    k_fill<<<gN, T, 0, stream>>>(deg, 1.0f, N);
    k_deg_accum<<<gE, T, 0, stream>>>(dstIdx, ew, deg, E);
    k_dinv<<<gN, T, 0, stream>>>(deg, dinv, N);

    // f16 operands for WMMA
    k_cvt_f16<<<gND, T, 0, stream>>>(x, xh, ND);
    k_cvt_wT<<<gW, T, 0, stream>>>(W1, w1t);
    k_cvt_wT<<<gW, T, 0, stream>>>(W2, w2t);

    // ---- layer 1 ----
    k_gemm_wmma<<<gTile, T, 0, stream>>>(xh, w1t, h, N);
    k_fill<<<gND, T, 0, stream>>>(accv, 0.0f, ND);
    k_scatter<<<gEW, T, 0, stream>>>(srcIdx, dstIdx, ew, dinv, h, accv, E);
    k_finalize_relu_f16<<<gND, T, 0, stream>>>(accv, h, dinv, b1, xh, ND);  // xh <- z (f16)

    // ---- layer 2 ----
    k_gemm_wmma<<<gTile, T, 0, stream>>>(xh, w2t, h, N);
    k_fill<<<gND, T, 0, stream>>>(accv, 0.0f, ND);
    k_scatter<<<gEW, T, 0, stream>>>(srcIdx, dstIdx, ew, dinv, h, accv, E);

    // ---- pooling + FC ----
    k_fill<<<(N_GRAPHS * DIM + T - 1) / T, T, 0, stream>>>(psum, 0.0f, N_GRAPHS * DIM);
    k_fill<<<1, T, 0, stream>>>(cnt, 0.0f, N_GRAPHS);
    k_count<<<gN, T, 0, stream>>>(batch, cnt, N);
    k_finalize_pool<<<gND, T, 0, stream>>>(accv, h, dinv, b2, batch, psum, ND);
    k_fc<<<(N_GRAPHS * DIM + T - 1) / T, T, 0, stream>>>(psum, cnt, Wfc, bfc, (float*)d_out);
}
